// AttEncoder_85418309583263
// MI455X (gfx1250) — matrix operand order repaired and checked by reading, not compiled, not run on hardware
//
#include <hip/hip_runtime.h>
#include <hip/hip_bf16.h>

typedef _Float16 f16;
typedef __attribute__((ext_vector_type(16))) _Float16 v16h;
typedef __attribute__((ext_vector_type(8)))  _Float16 half8;
typedef __attribute__((ext_vector_type(4)))  _Float16 half4;
typedef __attribute__((ext_vector_type(8)))  float    v8f;
typedef __attribute__((__vector_size__(16))) int      i32x4;

#define BB 8
#define SS 2048
#define DD 128
#define HH 4
#define DKK 32
#define LL 2
#define NROWS (BB*SS)          // 16384
#define WPAD (DD + 8)          // padded LDS row stride (halves) -> conflict-free B-frags
static __device__ __constant__ float kScale = 0.17677669529663687f; // 1/sqrt(32)
#define LN_EPS 1e-8f

#if defined(__has_builtin)
#if __has_builtin(__builtin_amdgcn_global_load_async_to_lds_b128)
#define HAVE_ASYNC_LDS 1
#endif
#endif

union Frag { v16h v; half8 h[2]; };

// 16-byte global -> LDS copy: async (ASYNCcnt-tracked) when available
__device__ __forceinline__ void copy16_to_lds(const void* gsrc, void* ldst) {
#if defined(HAVE_ASYNC_LDS)
  __builtin_amdgcn_global_load_async_to_lds_b128(
      (__attribute__((address_space(1))) i32x4*)gsrc,
      (__attribute__((address_space(3))) i32x4*)ldst, 0, 0);
#else
  *(half8*)ldst = *(const half8*)gsrc;
#endif
}
__device__ __forceinline__ void async_wait() {
#if defined(HAVE_ASYNC_LDS)
#if __has_builtin(__builtin_amdgcn_s_wait_asynccnt)
  __builtin_amdgcn_s_wait_asynccnt(0);
#else
  asm volatile("s_wait_asynccnt 0" ::: "memory");
#endif
#endif
}

__device__ __forceinline__ float red16_sum(float v) {
#pragma unroll
  for (int m = 8; m >= 1; m >>= 1) v += __shfl_xor(v, m, 32);
  return v;
}
__device__ __forceinline__ float red16_max(float v) {
#pragma unroll
  for (int m = 8; m >= 1; m >>= 1) v = fmaxf(v, __shfl_xor(v, m, 32));
  return v;
}
__device__ __forceinline__ float red32_sum(float v) {
#pragma unroll
  for (int m = 16; m >= 1; m >>= 1) v += __shfl_xor(v, m, 32);
  return v;
}

// ---------------------------------------------------------------- keep mask
__global__ void k_mask(const int* __restrict__ logs, const float* __restrict__ seqs,
                       float* __restrict__ x, float* __restrict__ keepf, int n) {
  int i = blockIdx.x * blockDim.x + threadIdx.x;
  if (i >= n) return;
  int row = i >> 7;
  float kp = (logs[row] != 0) ? 1.f : 0.f;
  x[i] = seqs[i] * kp;
  if ((i & 127) == 0) keepf[row] = kp;
}

// ------------------------------------------------- weights: f32[in][out] -> f16[out][in]
__global__ void k_wt(const float* __restrict__ src, f16* __restrict__ dst, int nmat) {
  int i = blockIdx.x * blockDim.x + threadIdx.x;
  int tot = nmat * DD * DD;
  if (i >= tot) return;
  int m = i / (DD * DD);
  int r = i - m * DD * DD;
  int o = r >> 7, in = r & 127;
  dst[i] = (f16)src[m * DD * DD + in * DD + o];
}

// ---------------------------------------------------------------- LayerNorm (fused)
// one wave per row. t = xin (+ add). y = LN(t,g,b) (* keep if keepf).
// writes: yf (f32, optional), yh (f16 of y, optional), xh (f16 of t pre-LN, optional)
__global__ __launch_bounds__(256) void k_ln(const float* __restrict__ xin,
                                            const float* __restrict__ add,
                                            const float* __restrict__ g,
                                            const float* __restrict__ b,
                                            float* __restrict__ yf,
                                            f16* __restrict__ yh,
                                            f16* __restrict__ xh,
                                            const float* __restrict__ keepf) {
  int wave = threadIdx.x >> 5, lane = threadIdx.x & 31;
  int row = blockIdx.x * 8 + wave;
  int base = row * DD + lane * 4;
  float4 xv = *(const float4*)(xin + base);
  if (add) {
    float4 av = *(const float4*)(add + base);
    xv.x += av.x; xv.y += av.y; xv.z += av.z; xv.w += av.w;
  }
  if (xh) {
    half4 hx; hx.x = (f16)xv.x; hx.y = (f16)xv.y; hx.z = (f16)xv.z; hx.w = (f16)xv.w;
    *(half4*)(xh + base) = hx;
  }
  float mean = red32_sum(xv.x + xv.y + xv.z + xv.w) * (1.f / 128.f);
  float d0 = xv.x - mean, d1 = xv.y - mean, d2 = xv.z - mean, d3 = xv.w - mean;
  float var = red32_sum(d0 * d0 + d1 * d1 + d2 * d2 + d3 * d3) * (1.f / 128.f);
  float rstd = rsqrtf(var + LN_EPS);
  float4 gv = *(const float4*)(g + lane * 4);
  float4 bv = *(const float4*)(b + lane * 4);
  float kp = keepf ? keepf[row] : 1.f;
  float y0 = (d0 * rstd * gv.x + bv.x) * kp;
  float y1 = (d1 * rstd * gv.y + bv.y) * kp;
  float y2 = (d2 * rstd * gv.z + bv.z) * kp;
  float y3 = (d3 * rstd * gv.w + bv.w) * kp;
  if (yf) { float4 o; o.x = y0; o.y = y1; o.z = y2; o.w = y3; *(float4*)(yf + base) = o; }
  if (yh) {
    half4 hy; hy.x = (f16)y0; hy.y = (f16)y1; hy.z = (f16)y2; hy.w = (f16)y3;
    *(half4*)(yh + base) = hy;
  }
}

// ---------------------------------------------------------------- WMMA GEMM [M,128]@[128,128]
// Weights staged once per block into LDS via async-to-LDS (padded rows: the 16
// B-fragment lanes stride 272B = 68 dwords -> 4*n mod 64 banks, conflict-free).
// modes: 0=Q->qh[b][h][s][dk], 1=K->kh[b][h][s][dk], 2=V->vT[b][h][dk][s],
//        3=FFN1: relu -> f16 [M][128], 4=FFN2: (resid+acc)*keep -> f32 [M][128]
__global__ __launch_bounds__(256) void k_gemm(const f16* __restrict__ A,
                                              const f16* __restrict__ WT,
                                              const float* __restrict__ bias,
                                              int mode,
                                              float* __restrict__ outF,
                                              f16* __restrict__ outH,
                                              const float* __restrict__ resid,
                                              const float* __restrict__ keepf) {
  __shared__ __align__(16) f16 Wlds[DD * WPAD];
  {
    int t = threadIdx.x;
    int row = t >> 1, part = t & 1;            // 2 threads per 128-half row
    const f16* src = WT + row * DD + part * 64;
    f16* dst = Wlds + row * WPAD + part * 64;
#pragma unroll
    for (int i = 0; i < 8; i++)                // 8 x 16B per thread
      copy16_to_lds(src + i * 8, dst + i * 8);
  }
  async_wait();
  __syncthreads();

  int wave = threadIdx.x >> 5, lane = threadIdx.x & 31;
  int rowbase = (blockIdx.x * 8 + wave) * 16;
  int hi = lane >> 4, n = lane & 15;

  // A fragments for K = 0..127 (4 chunks of 32); lane holds row rowbase+n
  Frag a[4];
  const f16* arow = A + (rowbase + n) * DD;
#pragma unroll
  for (int c = 0; c < 4; c++) {
    a[c].h[0] = *(const half8*)(arow + c * 32 + hi * 8);
    a[c].h[1] = *(const half8*)(arow + c * 32 + 16 + hi * 8);
  }
  int bidx = rowbase >> 11;           // batch
  int s0 = rowbase & (SS - 1);        // seq base

  for (int nt = 0; nt < 8; nt++) {
    int col = nt * 16 + n;
    float bv = bias[col];
    v8f acc;
#pragma unroll
    for (int r = 0; r < 8; r++) acc[r] = bv;
    const f16* wrow = Wlds + col * WPAD;
#pragma unroll
    for (int c = 0; c < 4; c++) {
      Frag bf;
      bf.h[0] = *(const half8*)(wrow + c * 32 + hi * 8);
      bf.h[1] = *(const half8*)(wrow + c * 32 + 16 + hi * 8);
      acc = __builtin_amdgcn_wmma_f32_16x16x32_f16(false, a[c].v, false, bf.v,
                                                   (short)0, acc, false, false);
    }
    if (mode <= 2) {                   // head split
      int hd = col >> 5, dk = col & 31;
#pragma unroll
      for (int r = 0; r < 8; r++) {
        int s = s0 + hi * 8 + r;
        if (mode == 2) outH[((bidx * HH + hd) * DKK + dk) * SS + s] = (f16)acc[r];
        else           outH[((bidx * HH + hd) * SS + s) * DKK + dk] = (f16)acc[r];
      }
    } else if (mode == 3) {            // relu -> f16
#pragma unroll
      for (int r = 0; r < 8; r++) {
        int row = rowbase + hi * 8 + r;
        outH[row * DD + col] = (f16)fmaxf(acc[r], 0.f);
      }
    } else {                           // residual + keep -> f32
#pragma unroll
      for (int r = 0; r < 8; r++) {
        int row = rowbase + hi * 8 + r;
        outF[row * DD + col] = (resid[row * DD + col] + acc[r]) * keepf[row];
      }
    }
  }
}

// ---------------------------------------------------------------- flash attention
// one wave per (b, head, 16-row q-tile); 32-key chunks; causal; online softmax.
__global__ __launch_bounds__(32) void k_attn(const f16* __restrict__ qh,
                                             const f16* __restrict__ kh,
                                             const f16* __restrict__ vT,
                                             float* __restrict__ out) {
  __shared__ __align__(16) f16 P[16 * 32];
  int lane = threadIdx.x & 31;
  int hi = lane >> 4, n = lane & 15;
  int qt = blockIdx.x, hd = blockIdx.y, b = blockIdx.z;
  int qbase = qt * 16;

  const f16* qptr = qh + ((b * HH + hd) * SS + qbase) * DKK;
  const f16* kptr = kh + ((b * HH + hd) * SS) * DKK;
  const f16* vptr = vT + ((b * HH + hd) * DKK) * SS;

  Frag qa;                              // Q tile 16x32 A-fragment, row = n
  qa.h[0] = *(const half8*)(qptr + n * DKK + hi * 8);
  qa.h[1] = *(const half8*)(qptr + n * DKK + 16 + hi * 8);

  v8f acc0 = {}, acc1 = {};
  float mrun[8], lrun[8];
#pragma unroll
  for (int r = 0; r < 8; r++) { mrun[r] = -INFINITY; lrun[r] = 0.f; }

  int nchunks = ((qbase + 15) >> 5) + 1;
  for (int ck = 0; ck < nchunks; ck++) {
    int kb = ck * 32;
    Frag kf0, kf1;                      // K^T B-fragments: lane n = key column
    kf0.h[0] = *(const half8*)(kptr + (kb + n) * DKK + hi * 8);
    kf0.h[1] = *(const half8*)(kptr + (kb + n) * DKK + 16 + hi * 8);
    kf1.h[0] = *(const half8*)(kptr + (kb + 16 + n) * DKK + hi * 8);
    kf1.h[1] = *(const half8*)(kptr + (kb + 16 + n) * DKK + 16 + hi * 8);
    // V^T B-fragments issued early: independent of scores/softmax, hides latency
    Frag vf0, vf1;
    vf0.h[0] = *(const half8*)(vptr + n * SS + kb + hi * 8);
    vf0.h[1] = *(const half8*)(vptr + n * SS + kb + 16 + hi * 8);
    vf1.h[0] = *(const half8*)(vptr + (16 + n) * SS + kb + hi * 8);
    vf1.h[1] = *(const half8*)(vptr + (16 + n) * SS + kb + 16 + hi * 8);
    if (ck + 1 < nchunks) {             // speculative prefetch of next chunk
      __builtin_prefetch(kptr + (kb + 32 + n) * DKK, 0, 0);
      __builtin_prefetch(vptr + n * SS + kb + 32, 0, 0);
    }
    v8f z = {};
    v8f sc0 = __builtin_amdgcn_wmma_f32_16x16x32_f16(false, qa.v, false, kf0.v,
                                                     (short)0, z, false, false);
    v8f sc1 = __builtin_amdgcn_wmma_f32_16x16x32_f16(false, qa.v, false, kf1.v,
                                                     (short)0, z, false, false);
    int c0 = kb + n, c1 = kb + 16 + n;
    float p0[8], p1[8];
#pragma unroll
    for (int r = 0; r < 8; r++) {
      int qg = qbase + hi * 8 + r;
      float t0 = (c0 <= qg) ? sc0[r] * kScale : -INFINITY;
      float t1 = (c1 <= qg) ? sc1[r] * kScale : -INFINITY;
      float mx = red16_max(fmaxf(t0, t1));
      float nm = fmaxf(mrun[r], mx);
      float alpha = __expf(mrun[r] - nm);
      float e0 = __expf(t0 - nm);
      float e1 = __expf(t1 - nm);
      lrun[r] = lrun[r] * alpha + red16_sum(e0 + e1);
      mrun[r] = nm;
      acc0[r] *= alpha; acc1[r] *= alpha;
      p0[r] = e0; p1[r] = e1;
    }
    // D-layout -> A-layout transpose via LDS
#pragma unroll
    for (int r = 0; r < 8; r++) {
      int m = hi * 8 + r;
      P[m * 32 + n]      = (f16)p0[r];
      P[m * 32 + 16 + n] = (f16)p1[r];
    }
    __syncthreads();
    Frag pa;
    pa.h[0] = *(const half8*)(&P[n * 32 + hi * 8]);
    pa.h[1] = *(const half8*)(&P[n * 32 + 16 + hi * 8]);
    __syncthreads();
    acc0 = __builtin_amdgcn_wmma_f32_16x16x32_f16(false, pa.v, false, vf0.v,
                                                  (short)0, acc0, false, false);
    acc1 = __builtin_amdgcn_wmma_f32_16x16x32_f16(false, pa.v, false, vf1.v,
                                                  (short)0, acc1, false, false);
  }
  float* obase = out + b * SS * DD + hd * DKK;
#pragma unroll
  for (int r = 0; r < 8; r++) {
    int s = qbase + hi * 8 + r;
    float inv = 1.f / lrun[r];
    obase[s * DD + n]      = acc0[r] * inv;
    obase[s * DD + 16 + n] = acc1[r] * inv;
  }
}

// ---------------------------------------------------------------- launch
extern "C" void kernel_launch(void* const* d_in, const int* in_sizes, int n_in,
                              void* d_out, int out_size, void* d_ws, size_t ws_size,
                              hipStream_t stream) {
  (void)in_sizes; (void)n_in; (void)out_size; (void)ws_size;
  const int*   logs = (const int*)d_in[0];
  const float* seqs = (const float*)d_in[1];
  const float* Wqkv = (const float*)d_in[2];
  const float* bqkv = (const float*)d_in[3];
  const float* ln1g = (const float*)d_in[4];
  const float* ln1b = (const float*)d_in[5];
  const float* ln2g = (const float*)d_in[6];
  const float* ln2b = (const float*)d_in[7];
  const float* W1   = (const float*)d_in[8];
  const float* b1   = (const float*)d_in[9];
  const float* W2   = (const float*)d_in[10];
  const float* b2   = (const float*)d_in[11];
  const float* lng  = (const float*)d_in[12];
  const float* lnb  = (const float*)d_in[13];
  float* outp = (float*)d_out;

  char* base = (char*)d_ws;
  size_t off = 0;
  auto carve = [&](size_t bytes) -> char* {
    char* p = base + off;
    off = (off + bytes + 255) & ~(size_t)255;
    return p;
  };
  const size_t NE = (size_t)BB * SS * DD;       // 2M elems
  float* xf    = (float*)carve(NE * 4);         // current x (f32)
  float* qin   = (float*)carve(NE * 4);         // LN1 output (residual base)
  float* attn  = (float*)carve(NE * 4);         // attention output
  float* keepf = (float*)carve((size_t)NROWS * 4);
  f16* xh    = (f16*)carve(NE * 2);             // f16 of x (K/V, FFN inputs)
  f16* qinh  = (f16*)carve(NE * 2);             // f16 of LN1 output (Q input)
  f16* qH    = (f16*)carve(NE * 2);             // [B,H,S,DK]
  f16* kH    = (f16*)carve(NE * 2);             // [B,H,S,DK]
  f16* vT    = (f16*)carve(NE * 2);             // [B,H,DK,S]
  f16* hh    = (f16*)carve(NE * 2);             // FFN hidden (f16)
  f16* WqkvT = (f16*)carve((size_t)LL * 3 * DD * DD * 2);
  f16* W1T   = (f16*)carve((size_t)LL * DD * DD * 2);
  f16* W2T   = (f16*)carve((size_t)LL * DD * DD * 2);

  k_mask<<<(unsigned)((NE + 255) / 256), 256, 0, stream>>>(logs, seqs, xf, keepf, (int)NE);
  k_wt<<<(LL * 3 * DD * DD + 255) / 256, 256, 0, stream>>>(Wqkv, WqkvT, LL * 3);
  k_wt<<<(LL * DD * DD + 255) / 256, 256, 0, stream>>>(W1, W1T, LL);
  k_wt<<<(LL * DD * DD + 255) / 256, 256, 0, stream>>>(W2, W2T, LL);

  dim3 gG(NROWS / 128), tG(256);
  int lnBlocks = NROWS / 8;
  for (int l = 0; l < LL; l++) {
    // q_in = LN1(x); also stash f16(x) and f16(q_in)
    k_ln<<<lnBlocks, 256, 0, stream>>>(xf, nullptr, ln1g + l * DD, ln1b + l * DD,
                                       qin, qinh, xh, nullptr);
    k_gemm<<<gG, tG, 0, stream>>>(qinh, WqkvT + (size_t)(l * 3 + 0) * DD * DD,
                                  bqkv + (l * 3 + 0) * DD, 0, nullptr, qH, nullptr, nullptr);
    k_gemm<<<gG, tG, 0, stream>>>(xh, WqkvT + (size_t)(l * 3 + 1) * DD * DD,
                                  bqkv + (l * 3 + 1) * DD, 1, nullptr, kH, nullptr, nullptr);
    k_gemm<<<gG, tG, 0, stream>>>(xh, WqkvT + (size_t)(l * 3 + 2) * DD * DD,
                                  bqkv + (l * 3 + 2) * DD, 2, nullptr, vT, nullptr, nullptr);
    k_attn<<<dim3(SS / 16, HH, BB), 32, 0, stream>>>(qH, kH, vT, attn);
    // x = LN2(q_in + attn); stash f16(x) for FFN
    k_ln<<<lnBlocks, 256, 0, stream>>>(qin, attn, ln2g + l * DD, ln2b + l * DD,
                                       xf, xh, nullptr, nullptr);
    k_gemm<<<gG, tG, 0, stream>>>(xh, W1T + (size_t)l * DD * DD, b1 + l * DD,
                                  3, nullptr, hh, nullptr, nullptr);
    // x = (x + hidden@W2 + b2) * keep
    k_gemm<<<gG, tG, 0, stream>>>(hh, W2T + (size_t)l * DD * DD, b2 + l * DD,
                                  4, xf, nullptr, xf, keepf);
  }
  // out = LN(x) * keep
  k_ln<<<lnBlocks, 256, 0, stream>>>(xf, nullptr, lng, lnb, outp, nullptr, nullptr, keepf);
}
